// Outer_Product_Mean_5806795784290
// MI455X (gfx1250) — matrix operand order repaired and checked by reading, not compiled, not run on hardware
//
#include <hip/hip_runtime.h>
#include <hip/hip_bf16.h>

typedef __attribute__((ext_vector_type(16))) _Float16 v16h;
typedef __attribute__((ext_vector_type(8)))  float    v8f;

#define L_SEQ    1024
#define IN_DIM   256
#define C_MSA    32
#define PAIR_DIM 64
#define NTOT     (L_SEQ * PAIR_DIM)   // 65536, flattened (j,p) = GEMM N dim

// ---------------------------------------------------------------------------
// A-fragment swizzle (v_wmma_f32_16x16x32_f16, 16x32 f16 A, ISA 7.12.2):
//  lane = m + 16*g  (m = row 0..15), 16 halfs/lane:
//   g=0: e0..7 -> K=0..7,   e8..15 -> K=16..23
//   g=1: e0..7 -> K=8..15,  e8..15 -> K=24..31
__device__ __forceinline__ void a_frag_pos(int k, int& g, int& e) {
    if      (k < 8)  { g = 0; e = k;      }
    else if (k < 16) { g = 1; e = k - 8;  }
    else if (k < 24) { g = 0; e = k - 8;  }
    else             { g = 1; e = k - 16; }
}

// ---------------------------------------------------------------------------
// Stage 1: s[i,c] = seq[i,:] . W1[c,:] + b1[c]   (f32 for stage 2)
// and scatter into fragment-order f16 A tiles: sA[(i/16)*512 + lane*16 + e]
__global__ void opm_proj1(const float* __restrict__ seq,
                          const float* __restrict__ W1,
                          const float* __restrict__ b1,
                          float* __restrict__ s32,
                          _Float16* __restrict__ sA) {
    __shared__ float row[IN_DIM];
    const int i = blockIdx.x;
    const int t = threadIdx.x;
    for (int k = t; k < IN_DIM; k += blockDim.x) row[k] = seq[i * IN_DIM + k];
    __syncthreads();
    if (t < C_MSA) {
        const float* w = W1 + t * IN_DIM;
        float acc = b1[t];
        #pragma unroll 8
        for (int k = 0; k < IN_DIM; ++k) acc += row[k] * w[k];
        s32[i * C_MSA + t] = acc;
        int g, e; a_frag_pos(t, g, e);
        const int m = i & 15, tile = i >> 4;
        sA[(size_t)tile * 512 + ((m + 16 * g) * 16 + e)] = (_Float16)acc;
    }
}

// ---------------------------------------------------------------------------
// Stage 2: tmp[j,p,d] = sum_c s[j,c] * W2[p, c*32 + d]
// Scattered directly into fragment-order f16 B tiles (32x16 f16 B layout:
//  lane = (n&15) + 16*((d>>3)&1), element e = (d&7) + 8*(d>>4), tile = n>>4, n = j*64+p):
__global__ void opm_proj2(const float* __restrict__ s32,
                          const float* __restrict__ W2,
                          _Float16* __restrict__ tmpB) {
    __shared__ float sj[C_MSA];
    const int j  = blockIdx.y;
    const int t  = threadIdx.x;                 // 0..255
    const int pd = blockIdx.x * 256 + t;        // 0..2047
    if (t < C_MSA) sj[t] = s32[j * C_MSA + t];
    __syncthreads();
    const int p = pd >> 5;
    const int d = pd & 31;
    const float* w = W2 + p * (C_MSA * C_MSA) + d;
    float acc = 0.f;
    #pragma unroll 8
    for (int c = 0; c < C_MSA; ++c) acc += sj[c] * w[c * C_MSA];
    const int n    = j * PAIR_DIM + p;
    const int lane = (n & 15) + 16 * ((d >> 3) & 1);
    const int e    = (d & 7) + 8 * (d >> 4);
    tmpB[(size_t)(n >> 4) * 512 + (lane * 16 + e)] = (_Float16)acc;
}

// ---------------------------------------------------------------------------
// Stage 3 (hot, HBM-bound): out[i, n] = A(16x32) x B(32x16) + pair + b2
// One wave -> 16 rows x 64 cols (one j, all p) = 4 WMMAs reusing A fragment.
// C/D f32 layout: lane l holds col n0+(l&15); vgpr v holds row i0 + (l>>4)*8 + v.
__global__ void __launch_bounds__(256)
opm_gemm(const _Float16* __restrict__ sA,
         const _Float16* __restrict__ tmpB,
         const float* __restrict__ pair,
         const float* __restrict__ b2,
         float* __restrict__ out) {
    const int wave  = blockIdx.x * 8 + (threadIdx.x >> 5);  // 0..65535
    const int lane  = threadIdx.x & 31;
    const int iTile = wave & 63;                 // 64 row tiles of 16
    const int j     = wave >> 6;                 // 0..1023
    const int g     = lane >> 4;
    const int ln    = lane & 15;
    const int i0    = iTile * 16;

    // A fragment: 32 contiguous bytes per lane (2x global_load_b128)
    v16h a = *(const v16h*)(sA + (size_t)iTile * 512 + lane * 16);

    const int nbase = j * PAIR_DIM;
    #pragma unroll
    for (int nb = 0; nb < 4; ++nb) {
        const int n0 = nbase + nb * 16;
        // B fragment: 32 contiguous bytes per lane, L2-resident
        v16h b = *(const v16h*)(tmpB + (size_t)(n0 >> 4) * 512 + lane * 16);

        // C = pair tile + b2 (streamed once: nontemporal)
        const float bias = b2[nb * 16 + ln];
        const float* pbase = pair + (size_t)(i0 + g * 8) * NTOT + n0 + ln;
        v8f c;
        #pragma unroll
        for (int v = 0; v < 8; ++v)
            c[v] = __builtin_nontemporal_load(pbase + (size_t)v * NTOT) + bias;

        // D = A x B + C  (f32 accumulate)
        v8f dacc = __builtin_amdgcn_wmma_f32_16x16x32_f16(
            /*neg_a=*/false, a, /*neg_b=*/false, b,
            /*c_mod=*/(short)0, c, /*reuse_a=*/false, /*reuse_b=*/false);

        float* obase = out + (size_t)(i0 + g * 8) * NTOT + n0 + ln;
        #pragma unroll
        for (int v = 0; v < 8; ++v)
            __builtin_nontemporal_store(dacc[v], obase + (size_t)v * NTOT);
    }
}

// ---------------------------------------------------------------------------
extern "C" void kernel_launch(void* const* d_in, const int* in_sizes, int n_in,
                              void* d_out, int out_size, void* d_ws, size_t ws_size,
                              hipStream_t stream) {
    (void)in_sizes; (void)n_in; (void)out_size; (void)ws_size;
    const float* seq  = (const float*)d_in[0];   // [1,1024,256]
    const float* pair = (const float*)d_in[1];   // [1,1024,1024,64]
    const float* W1   = (const float*)d_in[2];   // [32,256]
    const float* b1   = (const float*)d_in[3];   // [32]
    const float* W2   = (const float*)d_in[4];   // [64,1024]
    const float* b2   = (const float*)d_in[5];   // [64]
    float* out = (float*)d_out;

    char* ws = (char*)d_ws;
    float*    s32  = (float*)ws;                        // 128 KB
    _Float16* sA   = (_Float16*)(ws + (128 << 10));     //  64 KB, A fragments
    _Float16* tmpB = (_Float16*)(ws + (192 << 10));     //   4 MB, B fragments

    opm_proj1<<<L_SEQ, 64, 0, stream>>>(seq, W1, b1, s32, sA);
    opm_proj2<<<dim3((PAIR_DIM * C_MSA) / 256, L_SEQ), 256, 0, stream>>>(s32, W2, tmpB);
    // 64 iTiles * 1024 j = 65536 wave-tasks, 8 waves (256 thr) per block
    opm_gemm<<<65536 / 8, 256, 0, stream>>>(sA, tmpB, pair, b2, out);
}
